// VectorQuantizerEMA_13864154431886
// MI455X (gfx1250) — matrix operand order repaired and checked by reading, not compiled, not run on hardware
//
#include <hip/hip_runtime.h>
#include <hip/hip_bf16.h>

typedef __attribute__((ext_vector_type(16))) __bf16 v16bf;
typedef __attribute__((ext_vector_type(8)))  float  v8f;

#define NUM_EMB   2048
#define EMB_DIM   256
#define N_TOK     32768          // 32 * 32 * 32 tokens
#define HW        1024           // 32*32 spatial per batch image
#define TOK_BLK   128            // tokens per workgroup
#define B_ROW_DW  20             // 32 bf16 codes = 16 dwords, +4 pad (16B-aligned, bank-tiled)
#define B_BLK_DW  5120           // 256 rows * 20 dwords per staged B block
#define Q_ELEMS   8388608        // 32*256*32*32 quantized output elements

union Frag { uint4 q[2]; v16bf v; };

__device__ __forceinline__ unsigned short f2bf(float f) {
    unsigned int u = __float_as_uint(f);
    u += 0x7FFFu + ((u >> 16) & 1u);   // round-to-nearest-even
    return (unsigned short)(u >> 16);
}

// ---------------------------------------------------------------- init
__global__ void vq_init_kernel(int* __restrict__ hist, float* __restrict__ lossSum) {
    for (int k = threadIdx.x; k < NUM_EMB; k += 256) hist[k] = 0;
    if (threadIdx.x == 0) *lossSum = 0.f;
}

// ---------------------------------------------------------------- f32 -> bf16 codebook
__global__ void vq_cvt_kernel(const float* __restrict__ emb,
                              unsigned short* __restrict__ embBf) {
    const int e = (blockIdx.x * 256 + threadIdx.x) * 4;
    const float4 v = *(const float4*)(emb + e);
    ushort4 o;
    o.x = f2bf(v.x); o.y = f2bf(v.y); o.z = f2bf(v.z); o.w = f2bf(v.w);
    *(ushort4*)(embBf + e) = o;
}

// ---------------------------------------------------------------- 0.5*||e_k||^2 (exact f32)
__global__ void vq_norm_kernel(const float* __restrict__ emb,
                               float* __restrict__ halfNorm) {
    const int k = blockIdx.x * 256 + threadIdx.x;
    float s = 0.f;
    for (int d = 0; d < EMB_DIM; ++d) {
        const float v = emb[d * NUM_EMB + k];
        s += v * v;
    }
    halfNorm[k] = 0.5f * s;
}

// ---------------------------------------------------------------- main: WMMA scores + argmax
// 8 waves: waves 0-3 scan codes [0,1024), waves 4-7 scan [1024,2048).
// Each wave owns 32 tokens (two 16-row A-fragment sets in registers), so every
// B fragment (16 u16 LDS reads) feeds TWO wmmas (ds:wmma = 8:1, two independent acc chains).
__global__ __launch_bounds__(256)
void vq_argmin_kernel(const float* __restrict__ inp,
                      const unsigned short* __restrict__ embBf,
                      const float* __restrict__ halfNorm,
                      float* __restrict__ outIdxF,
                      int* __restrict__ hist) {
    // 64KB LDS, aliased: A-stage (128 tok x 256 d bf16) -> two B blocks -> merge scratch
    __shared__ __align__(16) unsigned short sh[32768];
    unsigned int* shDW = (unsigned int*)sh;

    const int t   = threadIdx.x;
    const int n0  = blockIdx.x * TOK_BLK;
    const int b   = n0 >> 10;
    const int hw0 = n0 & (HW - 1);
    const float* inBase = inp + (size_t)b * (EMB_DIM * HW) + hw0;

    // ---- stage A: transpose (d-strided global) -> token-major bf16 rows of 256
    {
        const int tok4  = (t & 31) * 4;
        const int dBase = t >> 5;
        for (int d = dBase; d < EMB_DIM; d += 8) {
            const float4 v = *(const float4*)(inBase + d * HW + tok4);
            sh[(tok4 + 0) * EMB_DIM + d] = f2bf(v.x);
            sh[(tok4 + 1) * EMB_DIM + d] = f2bf(v.y);
            sh[(tok4 + 2) * EMB_DIM + d] = f2bf(v.z);
            sh[(tok4 + 3) * EMB_DIM + d] = f2bf(v.w);
        }
    }
    __syncthreads();

    const int w    = t >> 5;         // wave 0..7
    const int L    = t & 31;         // lane
    const int half = w >> 2;         // code half: 0 or 1
    const int wl   = w & 3;          // wave-in-half: owns tokens [32wl, 32wl+32)
    const int kb   = (L >> 4) * 8;   // A-frag K base per ISA layout
    const int mA   = wl * 32 + (L & 15);

    // ---- hoist A fragments for this wave's 32 tokens into registers (reused 1024x)
    Frag aF0[8], aF1[8];
    #pragma unroll
    for (int kt = 0; kt < 8; ++kt) {
        const unsigned short* p0 = sh + mA * EMB_DIM + kt * 32 + kb;
        const unsigned short* p1 = p0 + 16 * EMB_DIM;
        aF0[kt].q[0] = *(const uint4*)(p0);
        aF0[kt].q[1] = *(const uint4*)(p0 + 16);
        aF1[kt].q[0] = *(const uint4*)(p1);
        aF1[kt].q[1] = *(const uint4*)(p1 + 16);
    }
    __syncthreads();   // A region dead; reuse LDS for B blocks

    float best[2][8];
    int   bidx[2][8];
    #pragma unroll
    for (int s = 0; s < 2; ++s)
        #pragma unroll
        for (int r = 0; r < 8; ++r) { best[s][r] = -3.4e38f; bidx[s][r] = 0; }

    const unsigned int* bbase = shDW + half * B_BLK_DW;

    for (int it = 0; it < 32; ++it) {
        // ---- stage two 32-code B blocks (one per code half): 2 x 20KB
        #pragma unroll
        for (int i = 0; i < 8; ++i) {
            const int lin = i * 256 + t;      // 2048 x 16B chunks
            const int blk = lin >> 10;
            const int rem = lin & 1023;
            const int row = rem >> 2;         // 4 chunks per 32-code row
            const int ch  = rem & 3;
            const uint4 v = *(const uint4*)(embBf + row * NUM_EMB +
                                            blk * 1024 + it * 32 + ch * 8);
            *(uint4*)(shDW + blk * B_BLK_DW + row * B_ROW_DW + ch * 4) = v;
        }
        __syncthreads();

        #pragma unroll
        for (int nt = 0; nt < 2; ++nt) {      // 2 x 16-code tiles per block
            v8f acc0 = {0.f, 0.f, 0.f, 0.f, 0.f, 0.f, 0.f, 0.f};
            v8f acc1 = {0.f, 0.f, 0.f, 0.f, 0.f, 0.f, 0.f, 0.f};
            #pragma unroll
            for (int kt = 0; kt < 8; ++kt) {  // D = 256 = 8 x 32
                Frag bF;                      // lane L holds B column N=L%16, 16 K values (table order)
                const unsigned short* p = (const unsigned short*)bbase + (kt * 32 + kb) * (2 * B_ROW_DW) + nt * 16 + (L & 15);
                #pragma unroll
                for (int s = 0; s < 8; ++s) {
                    bF.v[s]     = __builtin_bit_cast(__bf16, p[s * (2 * B_ROW_DW)]);
                    bF.v[8 + s] = __builtin_bit_cast(__bf16, p[(16 + s) * (2 * B_ROW_DW)]);
                }
                acc0 = __builtin_amdgcn_wmma_f32_16x16x32_bf16(
                         false, aF0[kt].v, false, bF.v, (short)0, acc0, false, false);
                acc1 = __builtin_amdgcn_wmma_f32_16x16x32_bf16(
                         false, aF1[kt].v, false, bF.v, (short)0, acc1, false, false);
            }
            const int   code = half * 1024 + it * 32 + nt * 16 + (L & 15);
            const float hn   = halfNorm[code];
            #pragma unroll
            for (int r = 0; r < 8; ++r) {     // score = x.e - 0.5||e||^2, branchless argmax
                const float s0 = acc0[r] - hn;
                const bool  g0 = s0 > best[0][r];
                bidx[0][r] = g0 ? code : bidx[0][r];
                best[0][r] = g0 ? s0 : best[0][r];
                const float s1 = acc1[r] - hn;
                const bool  g1 = s1 > best[1][r];
                bidx[1][r] = g1 ? code : bidx[1][r];
                best[1][r] = g1 ? s1 : best[1][r];
            }
        }
        __syncthreads();
    }

    // ---- argmax reduction across the 16 lanes sharing each token row
    float* mS = (float*)shDW;           // [2 halves][128 tokens] scores
    int*   mI = (int*)(shDW + 256);     // [2 halves][128 tokens] indices
    #pragma unroll
    for (int s = 0; s < 2; ++s) {
        #pragma unroll
        for (int r = 0; r < 8; ++r) {
            float v = best[s][r];
            int  ix = bidx[s][r];
            #pragma unroll
            for (int msk = 8; msk >= 1; msk >>= 1) {
                const float ov = __shfl_xor(v, msk, 32);
                const int   oi = __shfl_xor(ix, msk, 32);
                const bool  g  = (ov > v) || (ov == v && oi < ix);  // ties -> lowest idx
                ix = g ? oi : ix;
                v  = g ? ov : v;
            }
            if ((L & 15) == 0) {
                const int tl = wl * 32 + s * 16 + r + ((L >> 4) << 3);
                mS[half * TOK_BLK + tl] = v;
                mI[half * TOK_BLK + tl] = ix;
            }
        }
    }
    __syncthreads();

    // ---- merge the two code halves (tie -> half 0, which holds lower indices)
    if (t < TOK_BLK) {
        const float s0 = mS[t], s1 = mS[TOK_BLK + t];
        const int   i0 = mI[t], i1 = mI[TOK_BLK + t];
        const int   fin = (s1 > s0) ? i1 : i0;
        outIdxF[n0 + t] = (float)fin;
        atomicAdd(&hist[fin], 1);
    }
}

// ---------------------------------------------------------------- gather quantized + loss
__global__ __launch_bounds__(256)
void vq_gather_loss_kernel(const float* __restrict__ inp,
                           const float* __restrict__ emb,
                           const float* __restrict__ idxF,
                           float* __restrict__ outQ,
                           float* __restrict__ lossSum) {
    const int gid = blockIdx.x * 256 + threadIdx.x;
    const int e   = gid * 4;                 // 4 consecutive (same b,d; hw..hw+3)
    const int hw  = e & (HW - 1);
    const int bd  = e >> 10;
    const int d   = bd & (EMB_DIM - 1);
    const int bb  = bd >> 8;
    const int tok = (bb << 10) + hw;

    const float4 x = *(const float4*)(inp + e);
    const float* erow = emb + (size_t)d * NUM_EMB;
    float4 q;
    q.x = erow[(int)idxF[tok + 0]];
    q.y = erow[(int)idxF[tok + 1]];
    q.z = erow[(int)idxF[tok + 2]];
    q.w = erow[(int)idxF[tok + 3]];
    *(float4*)(outQ + e) = q;                // straight-through output == quantized

    float s = (q.x - x.x) * (q.x - x.x) + (q.y - x.y) * (q.y - x.y)
            + (q.z - x.z) * (q.z - x.z) + (q.w - x.w) * (q.w - x.w);
    #pragma unroll
    for (int msk = 16; msk >= 1; msk >>= 1) s += __shfl_xor(s, msk, 32);
    if ((threadIdx.x & 31) == 0) atomicAdd(lossSum, s);
}

// ---------------------------------------------------------------- loss + perplexity
__global__ void vq_finalize_kernel(const int* __restrict__ hist,
                                   const float* __restrict__ lossSum,
                                   float* __restrict__ outScalars) {
    __shared__ float red[8];
    const int t = threadIdx.x;
    float s = 0.f;
    for (int k = t; k < NUM_EMB; k += 256) {
        const float avg = (float)hist[k] * (1.0f / (float)N_TOK);
        s += avg * logf(avg + 1e-10f);
    }
    #pragma unroll
    for (int msk = 16; msk >= 1; msk >>= 1) s += __shfl_xor(s, msk, 32);
    if ((t & 31) == 0) red[t >> 5] = s;
    __syncthreads();
    if (t == 0) {
        float tot = 0.f;
        for (int i = 0; i < 8; ++i) tot += red[i];
        outScalars[0] = 1.25f * lossSum[0] / (float)Q_ELEMS;  // q_loss + 0.25*e_loss
        outScalars[1] = expf(-tot);                           // perplexity
    }
}

// ---------------------------------------------------------------- launcher
extern "C" void kernel_launch(void* const* d_in, const int* in_sizes, int n_in,
                              void* d_out, int out_size, void* d_ws, size_t ws_size,
                              hipStream_t stream) {
    (void)in_sizes; (void)n_in; (void)out_size; (void)ws_size;
    const float* inp = (const float*)d_in[0];   // (32, 256, 32, 32) f32
    const float* emb = (const float*)d_in[1];   // (256, 2048) f32
    float* out = (float*)d_out;

    char* ws = (char*)d_ws;
    unsigned short* embBf   = (unsigned short*)ws;                  // 1 MB bf16 codebook
    float*          halfNorm = (float*)(ws + (1u << 20));           // 8 KB
    int*            hist     = (int*)(ws + (1u << 20) + 8192);      // 8 KB
    float*          lossSum  = (float*)(ws + (1u << 20) + 16384);   // 4 B

    float* outQ    = out;                 // 8388608 quantized_st
    float* outScal = out + Q_ELEMS;       // [loss, perplexity]
    float* outIdx  = out + Q_ELEMS + 2;   // 32768 indices (as float)

    vq_init_kernel<<<dim3(1), dim3(256), 0, stream>>>(hist, lossSum);
    vq_cvt_kernel<<<dim3(512), dim3(256), 0, stream>>>(emb, embBf);
    vq_norm_kernel<<<dim3(8), dim3(256), 0, stream>>>(emb, halfNorm);
    vq_argmin_kernel<<<dim3(N_TOK / TOK_BLK), dim3(256), 0, stream>>>(
        inp, embBf, halfNorm, outIdx, hist);
    vq_gather_loss_kernel<<<dim3(Q_ELEMS / 1024), dim3(256), 0, stream>>>(
        inp, emb, outIdx, outQ, lossSum);
    vq_finalize_kernel<<<dim3(1), dim3(256), 0, stream>>>(hist, lossSum, outScal);
}
